// GAT_norm_68032281969087
// MI455X (gfx1250) — matrix operand order repaired and checked by reading, not compiled, not run on hardware
//
#include <hip/hip_runtime.h>
#include <hip/hip_bf16.h>

// ---------------------------------------------------------------------------
// GAT layer (heads=1) + PairNorm + ReLU for MI455X (gfx1250, wave32, WMMA).
// GEMM h = dropout(x) @ W runs on v_wmma_f32_16x16x32_bf16; graph softmax /
// aggregation uses f32 global atomics (order-preserving uint max + f32 add).
// ---------------------------------------------------------------------------

#define D_HID 512
#define NEG_SLOPE 0.2f

typedef __attribute__((ext_vector_type(16))) __bf16 v16bf;
typedef __attribute__((ext_vector_type(8)))  __bf16 v8bf;
typedef __attribute__((ext_vector_type(8)))  float  v8f;

__device__ __forceinline__ unsigned gat_hash(unsigned x, unsigned seed) {
  x ^= seed + 0x9e3779b9u + (x << 6) + (x >> 2);
  x *= 0x85ebca6bu; x ^= x >> 13;
  x *= 0xc2b2ae35u; x ^= x >> 16;
  return x;
}

// Order-preserving float<->uint map so unsigned atomicMax == float max.
__device__ __forceinline__ unsigned gat_ford_enc(float f) {
  unsigned u = __float_as_uint(f);
  return (u & 0x80000000u) ? ~u : (u | 0x80000000u);
}
__device__ __forceinline__ float gat_ford_dec(unsigned u) {
  return __uint_as_float((u & 0x80000000u) ? (u & 0x7fffffffu) : ~u);
}

// ---------------- prep: feature dropout (p=0.5) + bf16 convert -------------
__global__ void gat_prep_x(const float* __restrict__ x, __bf16* __restrict__ xb, int ND) {
  int idx = blockIdx.x * blockDim.x + threadIdx.x;
  if (idx >= ND) return;
  unsigned h = gat_hash((unsigned)idx, 42u);
  float v = ((h & 0xFFFFu) < 32768u) ? x[idx] * 2.0f : 0.0f;  // keep/(1-p)
  xb[idx] = (__bf16)v;
}

// ---------------- prep: W transpose to [N][K] bf16 -------------------------
__global__ void gat_prep_w(const float* __restrict__ W, __bf16* __restrict__ wbT, int DD) {
  int idx = blockIdx.x * blockDim.x + threadIdx.x;
  if (idx >= DD) return;
  int n = idx / D_HID, k = idx - n * D_HID;       // wbT[n][k] = W[k][n]
  wbT[idx] = (__bf16)W[(size_t)k * D_HID + n];
}

// ---------------- init: out=0, segment max/sum, colsum, ssq ----------------
__global__ void gat_init(float* __restrict__ out, unsigned* __restrict__ mOrd,
                         float* __restrict__ sArr, float* __restrict__ colsum,
                         float* __restrict__ ssq, int N, int ND) {
  int idx = blockIdx.x * blockDim.x + threadIdx.x;
  if (idx < ND) out[idx] = 0.0f;
  if (idx < N) { mOrd[idx] = 0u; sArr[idx] = 0.0f; }   // 0u < encoding of any float
  if (idx < D_HID) colsum[idx] = 0.0f;
  if (idx == 0) ssq[0] = 0.0f;
}

// ---------------- GEMM: h = xb @ W  (wave -> 16x64 tile) -------------------
__global__ void __launch_bounds__(256) gat_gemm_wmma(
    const __bf16* __restrict__ xb, const __bf16* __restrict__ wbT,
    float* __restrict__ h, int nRowTiles) {
  const int lane = threadIdx.x & 31;
  const int gw   = blockIdx.x * 8 + (threadIdx.x >> 5);
  const int row_tile = gw >> 3;         // 8 strips of 64 cols cover D=512
  const int strip    = gw & 7;
  if (row_tile >= nRowTiles) return;    // wave-uniform: EXEC stays all-ones
  const int m0 = row_tile * 16;
  const int n0 = strip * 64;
  const int hlf = lane >> 4;            // 0: K 0..15 | 1: K 16..31
  const int mn  = lane & 15;

  const __bf16* aBase = xb  + (size_t)(m0 + mn) * D_HID + hlf * 16;
  const __bf16* bBase = wbT + (size_t)(n0 + mn) * D_HID + hlf * 16;

  v8f acc0 = {}, acc1 = {}, acc2 = {}, acc3 = {};
  union { v16bf v; v8bf half[2]; } a;

  for (int k0 = 0; k0 < D_HID; k0 += 32) {
    a.half[0] = *(const v8bf*)(aBase + k0);
    a.half[1] = *(const v8bf*)(aBase + k0 + 8);
    v16bf b0 = *(const v16bf*)(bBase + k0);
    v16bf b1 = *(const v16bf*)(bBase + (size_t)16 * D_HID + k0);
    v16bf b2 = *(const v16bf*)(bBase + (size_t)32 * D_HID + k0);
    v16bf b3 = *(const v16bf*)(bBase + (size_t)48 * D_HID + k0);
    acc0 = __builtin_amdgcn_wmma_f32_16x16x32_bf16(false, a.v, false, b0, (short)0, acc0, false, false);
    acc1 = __builtin_amdgcn_wmma_f32_16x16x32_bf16(false, a.v, false, b1, (short)0, acc1, false, false);
    acc2 = __builtin_amdgcn_wmma_f32_16x16x32_bf16(false, a.v, false, b2, (short)0, acc2, false, false);
    acc3 = __builtin_amdgcn_wmma_f32_16x16x32_bf16(false, a.v, false, b3, (short)0, acc3, false, false);
  }
  // C/D layout: VGPR v -> row m0+v (lanes 0-15) / m0+8+v (lanes 16-31), col lane&15
  float* o = h + (size_t)(m0 + (hlf ? 8 : 0)) * D_HID + n0 + mn;
#pragma unroll
  for (int v = 0; v < 8; ++v) {
    o[(size_t)v * D_HID + 0]  = acc0[v];
    o[(size_t)v * D_HID + 16] = acc1[v];
    o[(size_t)v * D_HID + 32] = acc2[v];
    o[(size_t)v * D_HID + 48] = acc3[v];
  }
}

// scalar tail for N % 16 rows (N=30000 -> unused, kept for generality)
__global__ void gat_gemm_tail(const __bf16* __restrict__ xb, const __bf16* __restrict__ wbT,
                              float* __restrict__ h, int rowStart, int N) {
  int idx = blockIdx.x * blockDim.x + threadIdx.x;
  int nTail = N - rowStart;
  if (idx >= nTail * D_HID) return;
  int r = rowStart + idx / D_HID;
  int c = idx - (idx / D_HID) * D_HID;
  const __bf16* xr = xb + (size_t)r * D_HID;
  const __bf16* wc = wbT + (size_t)c * D_HID;
  float acc = 0.0f;
  for (int k = 0; k < D_HID; ++k) acc += (float)xr[k] * (float)wc[k];
  h[(size_t)r * D_HID + c] = acc;
}

// ---------------- per-node attention dots: a_s = h.att_src, a_d ------------
__global__ void __launch_bounds__(256) gat_node_dots(
    const float* __restrict__ h, const float* __restrict__ att_src,
    const float* __restrict__ att_dst, float* __restrict__ a_s,
    float* __restrict__ a_d, int N) {
  int lane = threadIdx.x & 31;
  int row  = blockIdx.x * 8 + (threadIdx.x >> 5);
  if (row >= N) return;
  const float* hr = h + (size_t)row * D_HID;
  float ss = 0.0f, sd = 0.0f;
  for (int j = lane; j < D_HID; j += 32) {
    float v = hr[j];
    ss += v * att_src[j];
    sd += v * att_dst[j];
  }
  for (int off = 16; off; off >>= 1) {
    ss += __shfl_down(ss, off, 32);
    sd += __shfl_down(sd, off, 32);
  }
  if (lane == 0) { a_s[row] = ss; a_d[row] = sd; }
}

// ---------------- edge logits + segment max --------------------------------
__global__ void gat_edge_logit_max(const int* __restrict__ srcA, const int* __restrict__ dstA,
                                   int E, int N, const float* __restrict__ a_s,
                                   const float* __restrict__ a_d,
                                   float* __restrict__ eArr, unsigned* __restrict__ mOrd) {
  int t = blockIdx.x * blockDim.x + threadIdx.x;
  if (t >= E + N) return;
  int s = (t < E) ? srcA[t] : (t - E);      // self-loops appended
  int d = (t < E) ? dstA[t] : (t - E);
  float e = a_s[s] + a_d[d];
  e = (e > 0.0f) ? e : e * NEG_SLOPE;       // LeakyReLU
  eArr[t] = e;
  atomicMax(mOrd + d, gat_ford_enc(e));
}

// ---------------- exp(e - m[dst]) + segment sum ----------------------------
__global__ void gat_edge_exp_sum(const int* __restrict__ srcA, const int* __restrict__ dstA,
                                 int E, int N, float* __restrict__ eArr,
                                 const unsigned* __restrict__ mOrd, float* __restrict__ sArr) {
  int t = blockIdx.x * blockDim.x + threadIdx.x;
  if (t >= E + N) return;
  int d = (t < E) ? dstA[t] : (t - E);
  float m = gat_ford_dec(mOrd[d]);          // every node has a self-loop -> finite
  float ee = __expf(eArr[t] - m);
  eArr[t] = ee;
  atomicAdd(sArr + d, ee);
}

// ---------------- alpha + attn dropout (p=0.6) + weighted scatter ----------
__global__ void __launch_bounds__(256) gat_scatter(
    const int* __restrict__ srcA, const int* __restrict__ dstA, int E, int N,
    const float* __restrict__ eeArr, const float* __restrict__ sArr,
    const float* __restrict__ h, float* __restrict__ out) {
  int lane = threadIdx.x & 31;
  int t = blockIdx.x * 8 + (threadIdx.x >> 5);
  if (t >= E + N) return;
  unsigned hsh = gat_hash((unsigned)t, 1337u);
  if ((hsh & 0xFFFFu) >= 26214u) return;    // keep-prob 0.4 (wave-uniform exit)
  int s = (t < E) ? srcA[t] : (t - E);
  int d = (t < E) ? dstA[t] : (t - E);
  float alpha = (eeArr[t] / (sArr[d] + 1e-16f)) * 2.5f;   // /(1-0.6)
  const float* hs = h + (size_t)s * D_HID;
  float* od = out + (size_t)d * D_HID;
  __builtin_prefetch(hs + lane, 0, 0);      // global_prefetch_b8
  for (int j = lane; j < D_HID; j += 32)
    atomicAdd(od + j, alpha * hs[j]);
}

// ---------------- bias add + column sums -----------------------------------
__global__ void __launch_bounds__(256) gat_bias_colsum(
    float* __restrict__ out, const float* __restrict__ bias,
    float* __restrict__ colsum, int N, int rowsPerBlock) {
  int tid = threadIdx.x;                    // thread owns cols tid, tid+256
  float b0 = bias[tid], b1 = bias[tid + 256];
  float c0 = 0.0f, c1 = 0.0f;
  int r0 = blockIdx.x * rowsPerBlock;
  int r1 = (r0 + rowsPerBlock < N) ? r0 + rowsPerBlock : N;
  for (int r = r0; r < r1; ++r) {
    float* row = out + (size_t)r * D_HID;
    float v0 = row[tid] + b0;
    float v1 = row[tid + 256] + b1;
    row[tid] = v0; row[tid + 256] = v1;
    c0 += v0; c1 += v1;
  }
  atomicAdd(colsum + tid, c0);
  atomicAdd(colsum + tid + 256, c1);
}

__global__ void gat_colmean(float* __restrict__ colsum, int N) {
  int j = blockIdx.x * blockDim.x + threadIdx.x;
  if (j < D_HID) colsum[j] *= (1.0f / (float)N);   // in-place -> column mean
}

// ---------------- sum of squared centered rows -----------------------------
__global__ void __launch_bounds__(256) gat_ssq(
    const float* __restrict__ out, const float* __restrict__ colmean,
    float* __restrict__ ssq, int N) {
  int lane = threadIdx.x & 31;
  int row  = blockIdx.x * 8 + (threadIdx.x >> 5);
  if (row >= N) return;
  const float* r = out + (size_t)row * D_HID;
  float acc = 0.0f;
  for (int j = lane; j < D_HID; j += 32) {
    float v = r[j] - colmean[j];
    acc += v * v;
  }
  for (int off = 16; off; off >>= 1) acc += __shfl_down(acc, off, 32);
  if (lane == 0) atomicAdd(ssq, acc);
}

__global__ void gat_scale(const float* __restrict__ ssq, float* __restrict__ scale, int N) {
  if (threadIdx.x == 0 && blockIdx.x == 0)
    scale[0] = rsqrtf(1e-6f + ssq[0] / (float)N);
}

// ---------------- PairNorm finalize + ReLU ---------------------------------
__global__ void gat_final(float* __restrict__ out, const float* __restrict__ colmean,
                          const float* __restrict__ scale, int ND) {
  int idx = blockIdx.x * blockDim.x + threadIdx.x;
  if (idx >= ND) return;
  int col = idx & (D_HID - 1);
  float v = (out[idx] - colmean[col]) * scale[0];
  out[idx] = (v > 0.0f) ? v : 0.0f;
}

// ---------------------------------------------------------------------------
extern "C" void kernel_launch(void* const* d_in, const int* in_sizes, int n_in,
                              void* d_out, int out_size, void* d_ws, size_t ws_size,
                              hipStream_t stream) {
  const float* x       = (const float*)d_in[0];
  const int*   ei      = (const int*)  d_in[1];
  const float* W       = (const float*)d_in[2];
  const float* att_src = (const float*)d_in[3];
  const float* att_dst = (const float*)d_in[4];
  const float* bias    = (const float*)d_in[5];
  float* out = (float*)d_out;

  const int N  = in_sizes[0] / D_HID;
  const int E  = in_sizes[1] / 2;
  const int ND = N * D_HID;
  const int DD = D_HID * D_HID;
  const int T  = E + N;                 // edges incl. self-loops
  const int* srcA = ei;
  const int* dstA = ei + E;

  // workspace carve-up (256B aligned)
  char* ws = (char*)d_ws;
  size_t off = 0;
  auto carve = [&](size_t bytes) -> char* {
    off = (off + 255) & ~(size_t)255;
    char* p = ws + off; off += bytes; return p;
  };
  __bf16*   xb     = (__bf16*)  carve((size_t)ND * 2);
  __bf16*   wbT    = (__bf16*)  carve((size_t)DD * 2);
  float*    h      = (float*)   carve((size_t)ND * 4);
  float*    a_s    = (float*)   carve((size_t)N * 4);
  float*    a_d    = (float*)   carve((size_t)N * 4);
  unsigned* mOrd   = (unsigned*)carve((size_t)N * 4);
  float*    sArr   = (float*)   carve((size_t)N * 4);
  float*    eArr   = (float*)   carve((size_t)T * 4);
  float*    colsum = (float*)   carve(D_HID * 4);
  float*    ssq    = (float*)   carve(4);
  float*    scale  = (float*)   carve(4);
  (void)ws_size; (void)n_in; (void)out_size;

  const int B = 256;
  // prep + init
  gat_prep_x<<<(ND + B - 1) / B, B, 0, stream>>>(x, xb, ND);
  gat_prep_w<<<(DD + B - 1) / B, B, 0, stream>>>(W, wbT, DD);
  gat_init  <<<(ND + B - 1) / B, B, 0, stream>>>(out, mOrd, sArr, colsum, ssq, N, ND);

  // GEMM: one wave per 16x64 tile; 8 strips span D=512
  const int nRowTiles = N / 16;
  if (nRowTiles > 0)
    gat_gemm_wmma<<<nRowTiles, B, 0, stream>>>(xb, wbT, h, nRowTiles);
  const int tailStart = nRowTiles * 16;
  if (tailStart < N)
    gat_gemm_tail<<<((N - tailStart) * D_HID + B - 1) / B, B, 0, stream>>>(xb, wbT, h, tailStart, N);

  // attention pipeline
  gat_node_dots     <<<(N + 7) / 8, B, 0, stream>>>(h, att_src, att_dst, a_s, a_d, N);
  gat_edge_logit_max<<<(T + B - 1) / B, B, 0, stream>>>(srcA, dstA, E, N, a_s, a_d, eArr, mOrd);
  gat_edge_exp_sum  <<<(T + B - 1) / B, B, 0, stream>>>(srcA, dstA, E, N, eArr, mOrd, sArr);
  gat_scatter       <<<(T + 7) / 8, B, 0, stream>>>(srcA, dstA, E, N, eArr, sArr, h, out);

  // bias + PairNorm + ReLU
  const int rowsPerBlock = 128;
  gat_bias_colsum<<<(N + rowsPerBlock - 1) / rowsPerBlock, B, 0, stream>>>(out, bias, colsum, N, rowsPerBlock);
  gat_colmean    <<<(D_HID + B - 1) / B, B, 0, stream>>>(colsum, N);
  gat_ssq        <<<(N + 7) / 8, B, 0, stream>>>(out, colsum, ssq, N);
  gat_scale      <<<1, 32, 0, stream>>>(ssq, scale, N);
  gat_final      <<<(ND + B - 1) / B, B, 0, stream>>>(out, colsum, scale, ND);
}
